// SharedWeightFC_57741540327891
// MI455X (gfx1250) — matrix operand order repaired and checked by reading, not compiled, not run on hardware
//
#include <hip/hip_runtime.h>

// ---------------------------------------------------------------------------
// SharedWeightFC for MI455X (gfx1250):
//   y_small[4096,512] = x[4096,4096] @ shared_weights[512,4096]^T  (bf16 WMMA, f32 acc)
//   y[4096,4096]      = tile(y_small, 8 along N) + bias
// ---------------------------------------------------------------------------

typedef __attribute__((ext_vector_type(16))) __bf16 v16bf;
typedef __attribute__((ext_vector_type(8)))  float  v8f;

#define BATCH  4096
#define IN_F   4096
#define SROWS  512
#define OUT_F  4096
#define REPS   8             // OUT_F / SROWS

#define BM     128
#define BN     128
#define BKO    64            // K per pipeline stage (2 x 32-deep WMMA slabs)
#define KSTEPS (IN_F / BKO)  // 64

// One 32-deep slab in WMMA register-image layout: [8 subtiles][32 lanes][8 dwords]
#define TILE_DW (8 * 32 * 8)  // 2048 dwords = 8 KB (bf16)

__device__ __forceinline__ unsigned int pack2_bf16(float lo, float hi) {
  // f32 -> bf16 round-to-nearest-even, two values packed {hi,lo}.
  // (u>>16)&1 -> v_bfe_u32; u + bit + 0x7fff -> v_add3_u32;
  // high-half merge -> single v_perm_b32.
  unsigned int ul = __float_as_uint(lo);
  unsigned int uh = __float_as_uint(hi);
  ul += 0x7FFFu + ((ul >> 16) & 1u);
  uh += 0x7FFFu + ((uh >> 16) & 1u);
  return __builtin_amdgcn_perm(uh, ul, 0x07060302u);  // {uh[31:16], ul[31:16]}
}

__global__ __launch_bounds__(256, 1)
void swfc_wmma_kernel(const float* __restrict__ x,
                      const float* __restrict__ w,
                      const float* __restrict__ bias,
                      float* __restrict__ out) {
  // [buffer][k-slab][register image]
  __shared__ unsigned int ldsA[2][2][TILE_DW];
  __shared__ unsigned int ldsB[2][2][TILE_DW];

  const int tid  = threadIdx.x;
  const int lane = tid & 31;
  const int wave = tid >> 5;     // 0..7
  const int wm   = wave & 3;     // wave row  (4 x 32 rows)
  const int wn   = wave >> 2;    // wave col  (2 x 64 cols)

  const int blockM = blockIdx.x * BM;
  const int blockN = blockIdx.y * BN;

  // ---- producer indexing: wave s fills subtile s of both A and B slabs ----
  // A operand (16-bit A 16x32, ISA 7.12.2): lane holds M=lane%16;
  //   lanes 0-15: K in {0..7,16..23}; lanes 16-31: K in {8..15,24..31};
  //   dword d -> K base = (d>>2)*16 + (lane>=16 ? 8 : 0) + (d&3)*2.
  // B operand (16-bit B 32x16): lane holds N=lane%16; same K order as A:
  //   dword d -> K base = (d>>2)*16 + (lane>=16 ? 8 : 0) + (d&3)*2.
  const int khalfA = (lane >> 4) << 3;   // 0 or 8
  const int khalfB = (lane >> 4) << 4;   // 0 or 16
  int kOffA[8], kOffB[8];
#pragma unroll
  for (int d = 0; d < 8; ++d) {
    kOffA[d] = ((d & 3) << 1) + khalfA + ((d >> 2) << 4);
    kOffB[d] = ((d & 3) << 1) + khalfA + ((d >> 2) << 4);
  }

  const int rowA = blockM + wave * 16 + (lane & 15);   // row of x
  const int rowB = blockN + wave * 16 + (lane & 15);   // row of shared_weights
  const float* __restrict__ aRow = x + (size_t)rowA * IN_F;
  const float* __restrict__ bRow = w + (size_t)rowB * IN_F;

  const int dstOff = (wave * 32 + lane) * 8;   // dword offset of this thread's image slot

  float2 ra[2][8], rb[2][8];

  auto loadTile = [&](int kt) {
    const int k0 = kt * BKO;
#pragma unroll
    for (int s = 0; s < 2; ++s) {
      const int kk = k0 + s * 32;
#pragma unroll
      for (int d = 0; d < 8; ++d) {
        ra[s][d] = *(const float2*)(aRow + kk + kOffA[d]);
        rb[s][d] = *(const float2*)(bRow + kk + kOffB[d]);
      }
    }
    if (kt + 1 < KSTEPS) {          // L2 prefetch hint for the next stage
      __builtin_prefetch(aRow + (kt + 1) * BKO, 0, 1);
      __builtin_prefetch(bRow + (kt + 1) * BKO, 0, 1);
    }
  };

  auto storeTile = [&](int buf) {
#pragma unroll
    for (int s = 0; s < 2; ++s) {
      unsigned int pa[8], pb[8];
#pragma unroll
      for (int d = 0; d < 8; ++d) {
        pa[d] = pack2_bf16(ra[s][d].x, ra[s][d].y);
        pb[d] = pack2_bf16(rb[s][d].x, rb[s][d].y);
      }
      uint4* a4 = (uint4*)(&ldsA[buf][s][0] + dstOff);
      uint4* b4 = (uint4*)(&ldsB[buf][s][0] + dstOff);
      a4[0] = make_uint4(pa[0], pa[1], pa[2], pa[3]);
      a4[1] = make_uint4(pa[4], pa[5], pa[6], pa[7]);
      b4[0] = make_uint4(pb[0], pb[1], pb[2], pb[3]);
      b4[1] = make_uint4(pb[4], pb[5], pb[6], pb[7]);
    }
  };

  // ---- accumulators: wave tile 32x64 = 2x4 WMMA tiles ----
  v8f vzero = {};
  v8f acc[2][4];
#pragma unroll
  for (int i = 0; i < 2; ++i)
#pragma unroll
    for (int j = 0; j < 4; ++j) acc[i][j] = vzero;

  auto consume = [&](int buf) {
#pragma unroll
    for (int s = 0; s < 2; ++s) {
      const unsigned int* baseA = &ldsA[buf][s][0];
      const unsigned int* baseB = &ldsB[buf][s][0];
      v16bf aOp[2], bOp[4];
#pragma unroll
      for (int i = 0; i < 2; ++i)
        aOp[i] = *(const v16bf*)(baseA + ((wm * 2 + i) * 32 + lane) * 8);
#pragma unroll
      for (int j = 0; j < 4; ++j)
        bOp[j] = *(const v16bf*)(baseB + ((wn * 4 + j) * 32 + lane) * 8);
#pragma unroll
      for (int i = 0; i < 2; ++i)
#pragma unroll
        for (int j = 0; j < 4; ++j)
          acc[i][j] = __builtin_amdgcn_wmma_f32_16x16x32_bf16(
              false, aOp[i], false, bOp[j], (short)0, acc[i][j], false, false);
    }
  };

  // ---- software-pipelined main loop (double-buffered LDS) ----
  loadTile(0);
  storeTile(0);
  __syncthreads();
#pragma unroll 1
  for (int kt = 1; kt < KSTEPS; ++kt) {
    loadTile(kt);             // global loads for next stage in flight...
    consume((kt - 1) & 1);    // ...while 16 WMMAs chew on the current one
    storeTile(kt & 1);
    __syncthreads();
  }
  consume((KSTEPS - 1) & 1);

  // ---- epilogue: bias + write each 32x64 wave tile to all 8 column replicas.
  // C/D layout: VGPR e -> M = e (lanes 0-15) or e+8 (lanes 16-31); N = lane%16.
  // r-loop kept rolled to limit static code size; stores stay 64B-coalesced
  // per half-wave and are the mandatory 64 MB output traffic.
  const int halfM = (lane >> 4) << 3;
  const int nl    = lane & 15;
#pragma unroll 1
  for (int r = 0; r < REPS; ++r) {
#pragma unroll
    for (int j = 0; j < 4; ++j) {
      const int col = blockN + (wn * 4 + j) * 16 + nl + r * SROWS;
      const float bv = bias[col];
#pragma unroll
      for (int i = 0; i < 2; ++i) {
        const int row0 = blockM + (wm * 2 + i) * 16 + halfM;
        float* op = out + (size_t)row0 * OUT_F + col;
#pragma unroll
        for (int e = 0; e < 8; ++e)
          op[(size_t)e * OUT_F] = acc[i][j][e] + bv;
      }
    }
  }
}

extern "C" void kernel_launch(void* const* d_in, const int* in_sizes, int n_in,
                              void* d_out, int out_size, void* d_ws, size_t ws_size,
                              hipStream_t stream) {
  const float* x    = (const float*)d_in[0];
  const float* w    = (const float*)d_in[1];
  const float* bias = (const float*)d_in[2];
  float* out        = (float*)d_out;

  dim3 grid(BATCH / BM, SROWS / BN);   // (32, 4)
  dim3 block(256);                     // 8 wave32
  swfc_wmma_kernel<<<grid, block, 0, stream>>>(x, w, bias, out);
}
